// LlamaAttention_35399120453872
// MI455X (gfx1250) — matrix operand order repaired and checked by reading, not compiled, not run on hardware
//
#include <hip/hip_runtime.h>
#include <hip/hip_bf16.h>

// Problem dims
#define Bq   2
#define Sq   2048
#define Hq   2048
#define NHq  16
#define HDq  128

typedef __attribute__((ext_vector_type(16))) __bf16 v16bf;
typedef __attribute__((ext_vector_type(8)))  float  v8f;
typedef __attribute__((ext_vector_type(4)))  unsigned int u32x4;
typedef __attribute__((ext_vector_type(8)))  int          i32x8;
typedef __attribute__((ext_vector_type(4)))  int          i32x4;

#if __has_builtin(__builtin_amdgcn_tensor_load_to_lds) && \
    __has_builtin(__builtin_amdgcn_s_wait_tensorcnt)
#define HAVE_TDM 1
#else
#define HAVE_TDM 0
#endif

union Frag { v16bf bf; uint4 u4[2]; };

__device__ inline unsigned short f2bf(float f) {
  unsigned int u = __float_as_uint(f);
  u = u + 0x7FFFu + ((u >> 16) & 1u);           // round-to-nearest-even
  return (unsigned short)(u >> 16);
}
__device__ inline float bf2f(unsigned short h) {
  return __uint_as_float(((unsigned int)h) << 16);
}
__device__ inline v8f wmma_bf16(Frag a, Frag b, v8f c) {
  return __builtin_amdgcn_wmma_f32_16x16x32_bf16(false, a.bf, false, b.bf,
                                                 (short)0, c, false, false);
}

#if HAVE_TDM
// Issue a 2D TDM tile load (bf16 elements) per CDNA5 ISA.
// D# group0: count=1, lds_addr, global_addr[56:0], type=2 ("image").
// D# group1: data_size=1 (2B), tensor_dim0/1, tile_dim0/1, tensor_dim0_stride.
// This toolchain exposes the 6-arg builtin:
//   (u32x4 g0, i32x8 g1, i32x4 g2, i32x4 g3, i32x8 g4, i32 cpol)
__device__ inline void tdm_load_2d_bf16(const void* gaddr, unsigned lds_off,
                                        unsigned tensor_d0, unsigned tensor_d1,
                                        unsigned tile_d0, unsigned tile_d1,
                                        unsigned stride_d0) {
  unsigned long long ga = (unsigned long long)gaddr;
  u32x4 g0;
  g0[0] = 1u;                                        // count=1, user mode
  g0[1] = lds_off;                                   // LDS byte address
  g0[2] = (unsigned)(ga & 0xFFFFFFFFu);              // global_addr[31:0]
  g0[3] = (unsigned)((ga >> 32) & 0x01FFFFFFu)       // global_addr[56:32]
          | (2u << 30);                              // type=2
  i32x8 g1;
  g1[0] = (int)(1u << 16);                           // wg_mask=0, data_size=2B
  g1[1] = (int)((tensor_d0 & 0xFFFFu) << 16);        // tensor_dim0[15:0]
  g1[2] = (int)((tensor_d0 >> 16) | ((tensor_d1 & 0xFFFFu) << 16));
  g1[3] = (int)((tensor_d1 >> 16) | (tile_d0 << 16));// tile_dim0
  g1[4] = (int)(tile_d1 & 0xFFFFu);                  // tile_dim1, tile_dim2=0
  g1[5] = (int)stride_d0;                            // dim0 stride[31:0]
  g1[6] = 0;                                         // stride[47:32], d1stride lo
  g1[7] = 0;
  i32x4 g2 = {0, 0, 0, 0};                           // 2D: groups 2/3 unused
  i32x4 g3 = {0, 0, 0, 0};
  i32x8 g4 = {0, 0, 0, 0, 0, 0, 0, 0};
  __builtin_amdgcn_tensor_load_to_lds(g0, g1, g2, g3, g4, 0);
}
#endif

// ---------------------------------------------------------------- reductions
__global__ __launch_bounds__(256) void absmean_reduce(const float* __restrict__ w,
                                                      size_t n, float* out) {
  __shared__ float sm[256];
  float s = 0.0f;
  for (size_t i = (size_t)blockIdx.x * 256 + threadIdx.x; i < n;
       i += (size_t)gridDim.x * 256)
    s += fabsf(w[i]);
  sm[threadIdx.x] = s;
  __syncthreads();
  for (int o = 128; o > 0; o >>= 1) {
    if ((int)threadIdx.x < o) sm[threadIdx.x] += sm[threadIdx.x + o];
    __syncthreads();
  }
  if (threadIdx.x == 0) atomicAdd(out, sm[0]);
}

__global__ __launch_bounds__(256) void quantize_w(const float* __restrict__ w,
                                                  unsigned short* __restrict__ wq,
                                                  size_t n, const float* sumptr,
                                                  float inv_n) {
  size_t i = (size_t)blockIdx.x * 256 + threadIdx.x;
  if (i >= n) return;
  float denom = (*sumptr) * inv_n + 1e-5f;      // mean(|w|) + eps
  float q = rintf(w[i] / denom);                // ties-to-even like jnp.round
  q = fminf(1.0f, fmaxf(-1.0f, q));
  wq[i] = f2bf(q);                              // ternary: exact in bf16
}

__global__ __launch_bounds__(256) void cvt_f32_bf16(const float* __restrict__ x,
                                                    unsigned short* __restrict__ y,
                                                    size_t n) {
  size_t i = (size_t)blockIdx.x * 256 + threadIdx.x;
  if (i < n) y[i] = f2bf(x[i]);
}

// ------------------------------------------------------ tiled bf16 WMMA GEMM
// C[M][N] = clamp((A[M][K] @ Bw[N][K]^T) / 16, -128, 128)
// tile: 64(M) x 128(N) per block, 8 waves as 4x2, K-step 64,
// TDM double-buffered LDS staging (wave1: A-tile, wave0: B-tile).
__global__ __launch_bounds__(256) void gemm_bl(const unsigned short* __restrict__ A,
                                               const unsigned short* __restrict__ Bw,
                                               void* __restrict__ outp,
                                               int M, int N, int K, int out_bf16) {
  __shared__ __align__(128) unsigned short As[2][64][64];   // 2 x 8 KB
  __shared__ __align__(128) unsigned short Bs[2][128][64];  // 2 x 16 KB
  const int tid  = threadIdx.x;
  const int lane = tid & 31;
  const int wid  = tid >> 5;
  const int wr   = wid & 3;        // 4 row tiles of 16
  const int wc   = wid >> 2;       // 2 col slabs of 64
  const int half = lane >> 4;
  const int l15  = lane & 15;
  const int m0 = blockIdx.y * 64;
  const int n0 = blockIdx.x * 128;

  v8f acc[4] = {};

#if HAVE_TDM
  const unsigned ldsA[2] = {(unsigned)(size_t)(void*)&As[0][0][0],
                            (unsigned)(size_t)(void*)&As[1][0][0]};
  const unsigned ldsB[2] = {(unsigned)(size_t)(void*)&Bs[0][0][0],
                            (unsigned)(size_t)(void*)&Bs[1][0][0]};
  // prologue: DMA first tiles into buffer 0
  if (wid == 0)
    tdm_load_2d_bf16(Bw + (size_t)n0 * K, ldsB[0],
                     (unsigned)K, (unsigned)(N - n0), 64u, 128u, (unsigned)K);
  if (wid == 1)
    tdm_load_2d_bf16(A + (size_t)m0 * K, ldsA[0],
                     (unsigned)K, (unsigned)(M - m0), 64u, 64u, (unsigned)K);
  if (wid < 2) __builtin_amdgcn_s_wait_tensorcnt((short)0);
  __syncthreads();
#endif

  int p = 0;
  for (int kb = 0; kb < K; kb += 64) {
#if HAVE_TDM
    // issue DMA for the next K-slab into the other buffer
    if (kb + 64 < K) {
      if (wid == 0)
        tdm_load_2d_bf16(Bw + (size_t)n0 * K + kb + 64, ldsB[p ^ 1],
                         (unsigned)(K - kb - 64), (unsigned)(N - n0),
                         64u, 128u, (unsigned)K);
      if (wid == 1)
        tdm_load_2d_bf16(A + (size_t)m0 * K + kb + 64, ldsA[p ^ 1],
                         (unsigned)(K - kb - 64), (unsigned)(M - m0),
                         64u, 64u, (unsigned)K);
    }
#else
    { // manual staging: A 8KB = 256x2x16B, B 16KB = 256x4x16B
#pragma unroll
      for (int j = 0; j < 2; ++j) {
        int t = tid + j * 256;           // 512 chunks
        int row = t >> 3, ko = (t & 7) * 8;
        *(uint4*)&As[p][row][ko] =
            *(const uint4*)(A + (size_t)(m0 + row) * K + kb + ko);
      }
#pragma unroll
      for (int j = 0; j < 4; ++j) {
        int t = tid + j * 256;           // 1024 chunks
        int row = t >> 3, ko = (t & 7) * 8;
        *(uint4*)&Bs[p][row][ko] =
            *(const uint4*)(Bw + (size_t)(n0 + row) * K + kb + ko);
      }
    }
    __syncthreads();
#endif

    // compute on buffer p: 2 K-substeps x 4 N-tiles = 8 WMMAs per wave
    const unsigned short(*Ap)[64] = As[p];
    const unsigned short(*Bp)[64] = Bs[p];
    Frag afr[2];
#pragma unroll
    for (int sub = 0; sub < 2; ++sub) {
      const unsigned short* ar = &Ap[wr * 16 + l15][sub * 32];
      afr[sub].u4[0] = *(const uint4*)(ar + half * 8);
      afr[sub].u4[1] = *(const uint4*)(ar + 16 + half * 8);
    }
#pragma unroll
    for (int sub = 0; sub < 2; ++sub) {
      Frag bfr[4];                     // distinct regs -> batched ds_loads
#pragma unroll
      for (int nt = 0; nt < 4; ++nt) {
        const uint4* bp =
            (const uint4*)&Bp[wc * 64 + nt * 16 + l15][sub * 32 + half * 8];
        bfr[nt].u4[0] = bp[0];
        bfr[nt].u4[1] = bp[2];
      }
#pragma unroll
      for (int nt = 0; nt < 4; ++nt)
        acc[nt] = wmma_bf16(afr[sub], bfr[nt], acc[nt]);
    }
    __syncthreads();                   // all waves done reading buffer p

#if HAVE_TDM
    if (kb + 64 < K) {
      if (wid < 2) __builtin_amdgcn_s_wait_tensorcnt((short)0);
      __syncthreads();                 // next buffer published to all waves
    }
#endif
    p ^= 1;
  }

  // epilogue: C layout row = v + 8*half, col = l15
#pragma unroll
  for (int nt = 0; nt < 4; ++nt) {
    int col = n0 + wc * 64 + nt * 16 + l15;
#pragma unroll
    for (int v = 0; v < 8; ++v) {
      int row = m0 + wr * 16 + v + 8 * half;
      float y = acc[nt][v] * 0.0625f;                 // /scale*Qb = /16
      y = fminf(128.0f, fmaxf(-128.0f, y));
      if (out_bf16)
        ((unsigned short*)outp)[(size_t)row * N + col] = f2bf(y);
      else
        ((float*)outp)[(size_t)row * N + col] = y;
    }
  }
}

// --------------------------------------------------------- RoPE + repacking
// qkv bf16 [B*S][3H] -> Q[bh][s][d], K[bh][s][d], Vt[bh][d][s] (bf16)
__global__ __launch_bounds__(256) void rope_reorg(const unsigned short* __restrict__ qkv,
                                                  const float* __restrict__ rot,
                                                  unsigned short* __restrict__ Q,
                                                  unsigned short* __restrict__ K,
                                                  unsigned short* __restrict__ Vt) {
  int idx = blockIdx.x * 256 + threadIdx.x;     // < B*S*NH*HD = 2^23
  int d = idx & (HDq - 1);
  int h = (idx >> 7) & (NHq - 1);
  int s = (idx >> 11) & (Sq - 1);
  int b = idx >> 22;
  size_t rowbase = (size_t)(b * Sq + s) * (3 * Hq);
  int col = h * HDq + d;
  float cs = rot[s * HDq + d];
  float sn = rot[Sq * HDq + s * HDq + d];
  int   dp  = (d < 64) ? d + 64 : d - 64;
  float sgn = (d < 64) ? -1.0f : 1.0f;          // rotate_half sign
  float q  = bf2f(qkv[rowbase + col]);
  float qp = bf2f(qkv[rowbase + h * HDq + dp]);
  float k  = bf2f(qkv[rowbase + Hq + col]);
  float kp = bf2f(qkv[rowbase + Hq + h * HDq + dp]);
  float v  = bf2f(qkv[rowbase + 2 * Hq + col]);
  size_t bh = (size_t)(b * NHq + h);
  Q[(bh * Sq + s) * HDq + d]  = f2bf(q * cs + sgn * qp * sn);
  K[(bh * Sq + s) * HDq + d]  = f2bf(k * cs + sgn * kp * sn);
  Vt[(bh * HDq + d) * Sq + s] = f2bf(v);
}

// ------------------------------------------------------------ flash attention
// grid (S/128, B*NH), 8 waves/block, each wave owns a 16-query tile
__global__ __launch_bounds__(256) void flash_attn(const unsigned short* __restrict__ Qb,
                                                  const unsigned short* __restrict__ Kb,
                                                  const unsigned short* __restrict__ Vtb,
                                                  unsigned short* __restrict__ Yb) {
  __shared__ __align__(32) unsigned short Ps[8][16][32]; // per-wave P staging
  const int tid  = threadIdx.x;
  const int lane = tid & 31;
  const int wid  = tid >> 5;
  const int half = lane >> 4;
  const int l15  = lane & 15;
  const int bh = blockIdx.y;
  const int b  = bh >> 4;
  const int h  = bh & 15;
  const int q0 = blockIdx.x * 128 + wid * 16;

  const unsigned short* Qp = Qb  + ((size_t)bh * Sq + q0) * HDq;
  const unsigned short* Kp = Kb  + (size_t)bh * Sq * HDq;
  const unsigned short* Vp = Vtb + (size_t)bh * HDq * Sq;

  // Q fragments for all 4 K-steps of HD=128 (A-layout)
  Frag qf[4];
#pragma unroll
  for (int kk = 0; kk < 4; ++kk) {
    const unsigned short* qrow = Qp + (size_t)l15 * HDq + kk * 32;
    qf[kk].u4[0] = *(const uint4*)(qrow + half * 8);
    qf[kk].u4[1] = *(const uint4*)(qrow + 16 + half * 8);
  }

  v8f o[8] = {};
  float m_i[8], l_i[8];
#pragma unroll
  for (int v = 0; v < 8; ++v) { m_i[v] = -1e30f; l_i[v] = 0.0f; }
  const float sm_scale = 0.08838834764831845f;  // 1/sqrt(128)

  const int kend = q0 + 16;                     // causal: keys <= q0+15
  for (int kb = 0; kb < kend; kb += 32) {
    // S = Q @ K^T : two 16x16 tiles (keys kb..kb+15, kb+16..kb+31)
    v8f s0 = {}, s1 = {};
#pragma unroll
    for (int kk = 0; kk < 4; ++kk) {
      Frag b0, b1;
      const uint4* kr0 =
          (const uint4*)(Kp + (size_t)(kb + l15) * HDq + kk * 32 + half * 8);
      const uint4* kr1 =
          (const uint4*)(Kp + (size_t)(kb + 16 + l15) * HDq + kk * 32 + half * 8);
      b0.u4[0] = kr0[0]; b0.u4[1] = kr0[2];
      b1.u4[0] = kr1[0]; b1.u4[1] = kr1[2];
      s0 = wmma_bf16(qf[kk], b0, s0);
      s1 = wmma_bf16(qf[kk], b1, s1);
    }

    // online softmax; C layout: row = v+8*half, col = l15
#pragma unroll
    for (int v = 0; v < 8; ++v) {
      int qg = q0 + v + 8 * half;
      float x0 = s0[v] * sm_scale;
      float x1 = s1[v] * sm_scale;
      if (kb + l15 > qg)      x0 = -1e30f;
      if (kb + 16 + l15 > qg) x1 = -1e30f;
      float mx = fmaxf(x0, x1);
#pragma unroll
      for (int off = 1; off < 16; off <<= 1)
        mx = fmaxf(mx, __shfl_xor(mx, off, 32));
      float mnew  = fmaxf(m_i[v], mx);
      float alpha = __expf(m_i[v] - mnew);
      float p0 = __expf(x0 - mnew);
      float p1 = __expf(x1 - mnew);
      float ps = p0 + p1;
#pragma unroll
      for (int off = 1; off < 16; off <<= 1)
        ps += __shfl_xor(ps, off, 32);
      l_i[v] = l_i[v] * alpha + ps;
      m_i[v] = mnew;
#pragma unroll
      for (int nt = 0; nt < 8; ++nt) o[nt][v] *= alpha;
      // stage P (bf16) into per-wave LDS tile [16 q][32 k]
      Ps[wid][v + 8 * half][l15]      = f2bf(p0);
      Ps[wid][v + 8 * half][16 + l15] = f2bf(p1);
    }
    asm volatile("s_wait_dscnt 0" ::: "memory");  // stores visible to wave's loads

    // P as A-fragment (16x32), V^T columns as B-fragments: O += P @ V
    Frag pf;
    pf.u4[0] = *(const uint4*)&Ps[wid][l15][half * 8];
    pf.u4[1] = *(const uint4*)&Ps[wid][l15][16 + half * 8];
#pragma unroll
    for (int nt = 0; nt < 8; ++nt) {
      Frag bv;
      const uint4* vr =
          (const uint4*)(Vp + (size_t)(nt * 16 + l15) * Sq + kb + half * 8);
      bv.u4[0] = vr[0]; bv.u4[1] = vr[2];
      o[nt] = wmma_bf16(pf, bv, o[nt]);
    }
  }

  // normalize and write back to [B*S][H] bf16 for the out-projection
#pragma unroll
  for (int nt = 0; nt < 8; ++nt) {
    int d = nt * 16 + l15;
#pragma unroll
    for (int v = 0; v < 8; ++v) {
      int row = v + 8 * half;
      float y = o[nt][v] / l_i[v];
      Yb[(size_t)(b * Sq + q0 + row) * Hq + h * HDq + d] = f2bf(y);
    }
  }
}

// ------------------------------------------------------------------- launch
extern "C" void kernel_launch(void* const* d_in, const int* in_sizes, int n_in,
                              void* d_out, int out_size, void* d_ws, size_t ws_size,
                              hipStream_t stream) {
  const float* x      = (const float*)d_in[0];
  const float* rotary = (const float*)d_in[1];
  const float* w_qkv  = (const float*)d_in[2];
  const float* w_o    = (const float*)d_in[3];
  float* out = (float*)d_out;

  const size_t BS  = (size_t)Bq * Sq;             // 4096
  const size_t N1  = 3 * Hq;                      // 6144
  const size_t nWq = N1 * Hq;                     // 12582912
  const size_t nWo = (size_t)Hq * Hq;             // 4194304
  const size_t nX  = BS * Hq;                     // 8388608
  const size_t nQ  = (size_t)Bq * NHq * Sq * HDq; // 8388608

  char* ws = (char*)d_ws;
  float* sums = (float*)ws;
  size_t off = 256;
  unsigned short* Xbf = (unsigned short*)(ws + off); off += nX  * 2;
  unsigned short* Wq  = (unsigned short*)(ws + off); off += nWq * 2;
  unsigned short* Wo  = (unsigned short*)(ws + off); off += nWo * 2;
  unsigned short* QKV = (unsigned short*)(ws + off); off += BS * N1 * 2;
  unsigned short* Qr  = (unsigned short*)(ws + off); off += nQ * 2;
  unsigned short* Kr  = (unsigned short*)(ws + off); off += nQ * 2;
  unsigned short* Vt  = (unsigned short*)(ws + off); off += nQ * 2;
  unsigned short* Ybf = (unsigned short*)(ws + off); off += nX * 2;

  (void)hipMemsetAsync(sums, 0, 2 * sizeof(float), stream);
  absmean_reduce<<<1024, 256, 0, stream>>>(w_qkv, nWq, &sums[0]);
  absmean_reduce<<<1024, 256, 0, stream>>>(w_o,  nWo, &sums[1]);
  quantize_w<<<(int)((nWq + 255) / 256), 256, 0, stream>>>(w_qkv, Wq, nWq, &sums[0],
                                                           1.0f / (float)nWq);
  quantize_w<<<(int)((nWo + 255) / 256), 256, 0, stream>>>(w_o, Wo, nWo, &sums[1],
                                                           1.0f / (float)nWo);
  cvt_f32_bf16<<<(int)(nX / 256), 256, 0, stream>>>(x, Xbf, nX);

  // QKV projection: [4096 x 6144] = Xbf @ Wq^T, bf16 out with BitLinear clamp
  gemm_bl<<<dim3(6144 / 128, 4096 / 64), 256, 0, stream>>>(Xbf, Wq, QKV,
                                                           4096, 6144, 2048, 1);
  rope_reorg<<<(int)(nQ / 256), 256, 0, stream>>>(QKV, rotary, Qr, Kr, Vt);
  flash_attn<<<dim3(Sq / 128, Bq * NHq), 256, 0, stream>>>(Qr, Kr, Vt, Ybf);
  // out projection: [4096 x 2048] f32 out
  gemm_bl<<<dim3(2048 / 128, 4096 / 64), 256, 0, stream>>>(Ybf, Wo, out,
                                                           4096, 2048, 2048, 0);
}
